// TrajectoryStateEncoder_52793738003060
// MI455X (gfx1250) — matrix operand order repaired and checked by reading, not compiled, not run on hardware
//
#include <hip/hip_runtime.h>

#define H_DIM   512
#define T_STEPS 64
#define E_EDGES 200000
#define WN      786432   // 1536*512 elements per weight matrix

typedef __attribute__((ext_vector_type(16))) __bf16 v16bf;
typedef __attribute__((ext_vector_type(8)))  float  v8f;

__device__ __forceinline__ unsigned short f2bf(float f) {
    unsigned int u = __float_as_uint(f);
    u += 0x7FFFu + ((u >> 16) & 1u);          // round-to-nearest-even
    return (unsigned short)(u >> 16);
}

__device__ __forceinline__ float wave_sum(float v) {
#pragma unroll
    for (int off = 16; off > 0; off >>= 1) v += __shfl_xor(v, off, 32);
    return v;
}

// Build a 16-halfword WMMA fragment from two 16B-aligned chunks.
__device__ __forceinline__ v16bf load_frag(const unsigned short* p0,
                                           const unsigned short* p1) {
    union { struct { uint4 lo, hi; } q; v16bf v; } u;
    u.q.lo = *reinterpret_cast<const uint4*>(p0);
    u.q.hi = *reinterpret_cast<const uint4*>(p1);
    return u.v;
}

// Layernorm of one H=512 row held as 16 values per lane (cols lane+32*i),
// result written to dst[col] (dst may be LDS row or global row).
__device__ __forceinline__ void ln_write(const float val[16], int lane,
                                         const float* __restrict__ lw,
                                         const float* __restrict__ lb,
                                         float* __restrict__ dst) {
    float s = 0.f;
#pragma unroll
    for (int i = 0; i < 16; ++i) s += val[i];
    s = wave_sum(s);
    float mean = s * (1.0f / (float)H_DIM);
    float vs = 0.f;
#pragma unroll
    for (int i = 0; i < 16; ++i) { float d = val[i] - mean; vs += d * d; }
    vs = wave_sum(vs);
    float rs = rsqrtf(vs * (1.0f / (float)H_DIM) + 1e-5f);
#pragma unroll
    for (int i = 0; i < 16; ++i) {
        int c = lane + i * 32;
        dst[c] = (val[i] - mean) * rs * lw[c] + lb[c];
    }
}

__global__ void cvt_weights_bf16(const float* __restrict__ wi,
                                 const float* __restrict__ wh,
                                 unsigned short* __restrict__ oi,
                                 unsigned short* __restrict__ oh) {
    int base = (blockIdx.x * blockDim.x + threadIdx.x) * 4;
#pragma unroll
    for (int j = 0; j < 4; ++j) {
        int k = base + j;
        if (k < WN) { oi[k] = f2bf(wi[k]); oh[k] = f2bf(wh[k]); }
    }
}

__global__ __launch_bounds__(256)
void traj_gru_kernel(const int*   __restrict__ actions,
                     const float* __restrict__ edge_tokens,
                     const int*   __restrict__ stop_indices,
                     const float* __restrict__ qtok,
                     const float* __restrict__ node_tokens,
                     const int*   __restrict__ snl,
                     const int*   __restrict__ sptr,
                     const float* __restrict__ b_ih,
                     const float* __restrict__ b_hh,
                     const float* __restrict__ ln_w,
                     const float* __restrict__ ln_b,
                     const unsigned short* __restrict__ wih,
                     const unsigned short* __restrict__ whh,
                     float* __restrict__ out) {
    __shared__ float          hf[16][516];     // f32 hidden state (padded)
    __shared__ unsigned short xb[16][528];     // bf16 gathered edge tokens
    __shared__ unsigned short hb[16][528];     // bf16 mirror of h
    __shared__ int s_active[16];
    __shared__ int s_done[16];
    __shared__ int s_eidx[16];

    const int tid   = threadIdx.x;
    const int wave  = tid >> 5;
    const int lane  = tid & 31;
    const int gbase = blockIdx.x * 16;

    if (tid < 16) s_done[tid] = 0;

    // ---- h0 = LN(question + segment_mean(node_tokens[start])) ----
    for (int rr = 0; rr < 2; ++rr) {
        int m = wave * 2 + rr;
        int g = gbase + m;
        int p0 = sptr[g], p1 = sptr[g + 1];
        float acc[16];
#pragma unroll
        for (int i = 0; i < 16; ++i) acc[i] = 0.f;
        for (int e = p0; e < p1; ++e) {
            const float* row = node_tokens + (size_t)snl[e] * H_DIM;
#pragma unroll
            for (int i = 0; i < 16; ++i) acc[i] += row[lane + i * 32];
        }
        int cnt = p1 - p0; if (cnt < 1) cnt = 1;
        float inv = 1.f / (float)cnt;
        const float* q = qtok + (size_t)g * H_DIM;
        float val[16];
#pragma unroll
        for (int i = 0; i < 16; ++i) val[i] = q[lane + i * 32] + acc[i] * inv;
        ln_write(val, lane, ln_w, ln_b, &hf[m][0]);
    }
    __syncthreads();

    const int n    = lane & 15;      // N column within tile; also A row index
    const int m16  = lane >> 4;      // which half of the lane group
    const int ko   = m16 * 16;       // B-fragment K offset
    const int ao   = m16 * 8;        // unused; A fragments also load at ko

    for (int t = 0; t < T_STEPS; ++t) {
        // ---- (a) emit emb = LN(h) BEFORE the update ----
        for (int rr = 0; rr < 2; ++rr) {
            int m = wave * 2 + rr;
            int g = gbase + m;
            float val[16];
#pragma unroll
            for (int i = 0; i < 16; ++i) val[i] = hf[m][lane + i * 32];
            float* o = out + ((size_t)g * T_STEPS + t) * H_DIM;
            ln_write(val, lane, ln_w, ln_b, o);
        }
        // ---- per-graph control flags ----
        if (tid < 16) {
            int g = gbase + tid;
            int a = actions[(size_t)g * T_STEPS + t];
            int is_stop = (a == stop_indices[g]) || (a < 0);
            s_active[tid] = (!s_done[tid]) && (!is_stop);
            s_done[tid]  |= is_stop;
            int idx = a < 0 ? 0 : (a >= E_EDGES ? E_EDGES - 1 : a);
            s_eidx[tid] = idx;
        }
        __syncthreads();

        // ---- (b) gather edge rows -> bf16 LDS; mirror h -> bf16 ----
        for (int rr = 0; rr < 2; ++rr) {
            int m = wave * 2 + rr;
            const float4* row = reinterpret_cast<const float4*>(
                edge_tokens + (size_t)s_eidx[m] * H_DIM);
#pragma unroll
            for (int i = 0; i < 4; ++i) {
                int c4 = lane + i * 32;
                float4 v = row[c4];
                int c = c4 * 4;
                xb[m][c + 0] = f2bf(v.x); xb[m][c + 1] = f2bf(v.y);
                xb[m][c + 2] = f2bf(v.z); xb[m][c + 3] = f2bf(v.w);
            }
        }
#pragma unroll
        for (int j = 0; j < 32; ++j) {
            int flat = tid + j * 256;
            int m = flat >> 9, c = flat & 511;
            hb[m][c] = f2bf(hf[m][c]);
        }
        __syncthreads();

        // ---- (c) WMMA GEMMs + gates; each wave owns 4 column tiles ----
#pragma unroll 1
        for (int ti = 0; ti < 4; ++ti) {
            int jt  = wave * 4 + ti;
            int cb  = jt * 16;
            int col = cb + n;

            float br  = b_ih[col]             + b_hh[col];
            float bz  = b_ih[H_DIM + col]     + b_hh[H_DIM + col];
            float bin = b_ih[2 * H_DIM + col];
            float bhn = b_hh[2 * H_DIM + col];
            v8f aR  = {br, br, br, br, br, br, br, br};
            v8f aZ  = {bz, bz, bz, bz, bz, bz, bz, bz};
            v8f aIN = {bin, bin, bin, bin, bin, bin, bin, bin};
            v8f aHN = {bhn, bhn, bhn, bhn, bhn, bhn, bhn, bhn};

            const unsigned short* pIR = wih + (size_t)col * H_DIM;
            const unsigned short* pHR = whh + (size_t)col * H_DIM;
            const unsigned short* pIZ = wih + (size_t)(H_DIM + col) * H_DIM;
            const unsigned short* pHZ = whh + (size_t)(H_DIM + col) * H_DIM;
            const unsigned short* pIN = wih + (size_t)(2 * H_DIM + col) * H_DIM;
            const unsigned short* pHN = whh + (size_t)(2 * H_DIM + col) * H_DIM;
            const unsigned short* xrow = &xb[n][0];
            const unsigned short* hrow = &hb[n][0];

#pragma unroll 4
            for (int kk = 0; kk < 16; ++kk) {
                int kb = kk * 32;
                v16bf ax  = load_frag(xrow + kb + ko, xrow + kb + ko + 8);
                v16bf ah  = load_frag(hrow + kb + ko, hrow + kb + ko + 8);
                v16bf bIR = load_frag(pIR + kb + ko, pIR + kb + ko + 8);
                v16bf bIZ = load_frag(pIZ + kb + ko, pIZ + kb + ko + 8);
                v16bf bIN = load_frag(pIN + kb + ko, pIN + kb + ko + 8);
                v16bf bHR = load_frag(pHR + kb + ko, pHR + kb + ko + 8);
                v16bf bHZ = load_frag(pHZ + kb + ko, pHZ + kb + ko + 8);
                v16bf bHN = load_frag(pHN + kb + ko, pHN + kb + ko + 8);
                // group by shared A operand (HW can reuse A across consecutive WMMAs)
                aR  = __builtin_amdgcn_wmma_f32_16x16x32_bf16(false, ax, false, bIR, (short)0, aR,  false, false);
                aZ  = __builtin_amdgcn_wmma_f32_16x16x32_bf16(false, ax, false, bIZ, (short)0, aZ,  false, false);
                aIN = __builtin_amdgcn_wmma_f32_16x16x32_bf16(false, ax, false, bIN, (short)0, aIN, false, false);
                aR  = __builtin_amdgcn_wmma_f32_16x16x32_bf16(false, ah, false, bHR, (short)0, aR,  false, false);
                aZ  = __builtin_amdgcn_wmma_f32_16x16x32_bf16(false, ah, false, bHZ, (short)0, aZ,  false, false);
                aHN = __builtin_amdgcn_wmma_f32_16x16x32_bf16(false, ah, false, bHN, (short)0, aHN, false, false);
            }
            // gates + masked state update; this lane owns rows m16*8..m16*8+7 of col
#pragma unroll
            for (int v = 0; v < 8; ++v) {
                int m = v + m16 * 8;
                float r  = 1.f / (1.f + __expf(-aR[v]));
                float z  = 1.f / (1.f + __expf(-aZ[v]));
                float nn = tanhf(aIN[v] + r * aHN[v]);
                float hold = hf[m][col];
                float hnew = (1.f - z) * nn + z * hold;
                hf[m][col] = s_active[m] ? hnew : hold;
            }
        }
        __syncthreads();
    }
}

extern "C" void kernel_launch(void* const* d_in, const int* in_sizes, int n_in,
                              void* d_out, int out_size, void* d_ws, size_t ws_size,
                              hipStream_t stream) {
    const int*   actions      = (const int*)  d_in[0];
    const float* edge_tokens  = (const float*)d_in[1];
    const int*   stop_indices = (const int*)  d_in[2];
    const float* qtok         = (const float*)d_in[3];
    const float* node_tokens  = (const float*)d_in[4];
    const int*   snl          = (const int*)  d_in[5];
    const int*   sptr         = (const int*)  d_in[6];
    const float* w_ih         = (const float*)d_in[7];
    const float* w_hh         = (const float*)d_in[8];
    const float* b_ih         = (const float*)d_in[9];
    const float* b_hh         = (const float*)d_in[10];
    const float* ln_w         = (const float*)d_in[11];
    const float* ln_b         = (const float*)d_in[12];

    unsigned short* wih_bf = (unsigned short*)d_ws;
    unsigned short* whh_bf = wih_bf + WN;

    // One-time (per call, deterministic) weight down-conversion: 3 MB, L2-resident.
    cvt_weights_bf16<<<WN / 1024, 256, 0, stream>>>(w_ih, w_hh, wih_bf, whh_bf);

    // 1024 graphs / 16 per block = 64 blocks, 8 wave32s each.
    traj_gru_kernel<<<64, 256, 0, stream>>>(actions, edge_tokens, stop_indices,
                                            qtok, node_tokens, snl, sptr,
                                            b_ih, b_hh, ln_w, ln_b,
                                            wih_bf, whh_bf, (float*)d_out);
}
